// MHA_9895604650102
// MI455X (gfx1250) — matrix operand order repaired and checked by reading, not compiled, not run on hardware
//
#include <hip/hip_runtime.h>

typedef unsigned short ushort_t;
typedef __attribute__((ext_vector_type(16))) __bf16 v16bf;
typedef __attribute__((ext_vector_type(8)))  float  v8f;
typedef __attribute__((ext_vector_type(4)))  float  f32x4;

union ABPack { f32x4 f[2]; v16bf v; ushort_t u[16]; };

#define BB   2
#define SS   2048
#define EE   1024
#define HH   16
#define DD   64
#define NQKV 3072
#define MTOT (BB*SS)   /* 4096 */

static __device__ __forceinline__ ushort_t f2bf(float f) {
  union { float f; unsigned u; } c; c.f = f;
  unsigned u = c.u;
  return (ushort_t)((u + 0x7FFFu + ((u >> 16) & 1u)) >> 16);
}

static __device__ __forceinline__ v8f wmma_bf16(v16bf a, v16bf b, v8f c) {
  return __builtin_amdgcn_wmma_f32_16x16x32_bf16(false, a, false, b, (short)0, c,
                                                 false, false);
}

// ---------------------------------------------------------------------------
// Kernel 1: qkv = x @ Wqkv + b  (bf16 WMMA, fp32 accum), RoPE fused in epilogue.
// Writes q,k as [B,H,S,D] bf16 and v transposed as [B,H,D,S] bf16.
// ---------------------------------------------------------------------------
__global__ __launch_bounds__(128) void qkv_rope_kernel(
    const float* __restrict__ x, const float* __restrict__ Wqkv,
    const float* __restrict__ bqkv,
    ushort_t* __restrict__ qo, ushort_t* __restrict__ ko,
    ushort_t* __restrict__ vTo)
{
  __shared__ alignas(16) ushort_t sA[64][32];   // A tile, row-major [m][k]
  __shared__ alignas(16) ushort_t sBt[64][32];  // B tile transposed [n][k]
  const int tid = threadIdx.x, lane = tid & 31, wv = tid >> 5;
  const int lane15 = lane & 15, hi8 = (lane >> 4) << 3;
  const int m0 = blockIdx.x * 64, n0 = blockIdx.y * 64;

  v8f acc[4];
  for (int t = 0; t < 4; ++t)
    for (int r = 0; r < 8; ++r) acc[t][r] = 0.0f;

  const int kbA = (lane < 16) ? 0 : 16;
  const int kbB = (lane < 16) ? 0 : 16;

  for (int k0 = 0; k0 < EE; k0 += 32) {
    for (int e = tid; e < 64 * 32; e += 128) {
      int r = e >> 5, c = e & 31;
      sA[r][c] = f2bf(x[(size_t)(m0 + r) * EE + k0 + c]);
    }
    for (int e = tid; e < 32 * 64; e += 128) {
      int kr = e >> 6, nc = e & 63;
      sBt[nc][kr] = f2bf(Wqkv[(size_t)(k0 + kr) * NQKV + n0 + nc]);
    }
    __syncthreads();
    if (k0 + 32 < EE) {                       // prefetch next tiles
      __builtin_prefetch(&x[(size_t)(m0 + (tid >> 1)) * EE + k0 + 32], 0, 0);
      __builtin_prefetch(&Wqkv[(size_t)(k0 + 32 + (tid >> 2)) * NQKV + n0], 0, 0);
    }
    ABPack a;
    const f32x4* rowA = (const f32x4*)&sA[16 * wv + lane15][0];
    a.f[0] = rowA[kbA >> 3];
    a.f[1] = rowA[(kbA >> 3) + 1];
    for (int t = 0; t < 4; ++t) {
      ABPack bfr;
      const f32x4* rowB = (const f32x4*)&sBt[16 * t + lane15][0];
      bfr.f[0] = rowB[kbB >> 3];
      bfr.f[1] = rowB[(kbB >> 3) + 1];
      acc[t] = wmma_bf16(a.v, bfr.v, acc[t]);
    }
    __syncthreads();
  }

  // Epilogue: bias + RoPE (pairs live in tiles t and t+2, same lane/register).
  const int which = n0 >> 10;                 // 0=q, 1=k, 2=v
  const int head  = (n0 >> 6) & (HH - 1);
  float bias[4];
  for (int t = 0; t < 4; ++t) bias[t] = bqkv[n0 + 16 * t + lane15];
  const float KROPE = 0.28782313662425572f;   // ln(10000)/32

  for (int r = 0; r < 8; ++r) {
    const int g  = m0 + 16 * wv + r + hi8;    // global row in [0,4096)
    const int bi = g >> 11;                   // batch
    const int sp = g & (SS - 1);              // position
    if (which < 2) {
      ushort_t* dst = (which == 0) ? qo : ko;
      const size_t base = (((size_t)bi * HH + head) * SS + sp) * DD;
      for (int t = 0; t < 2; ++t) {
        const int d = 16 * t + lane15;        // 0..31
        float x1 = acc[t][r]     + bias[t];
        float x2 = acc[t + 2][r] + bias[t + 2];
        float th = (float)sp * __expf(-(float)d * KROPE);
        float sn, cs;
        __sincosf(th, &sn, &cs);
        dst[base + d]      = f2bf(x1 * cs - x2 * sn);
        dst[base + d + 32] = f2bf(x1 * sn + x2 * cs);
      }
    } else {
      for (int t = 0; t < 4; ++t) {
        const int d = 16 * t + lane15;
        vTo[(((size_t)bi * HH + head) * DD + d) * SS + sp] =
            f2bf(acc[t][r] + bias[t]);
      }
    }
  }
}

// ---------------------------------------------------------------------------
// Kernel 2: causal flash attention. Workgroup = (b, h, 64 q-rows); each of the
// 4 waves owns a 16-row q fragment and streams 32-key blocks with online
// softmax. Scores and ctx both via bf16 WMMA.
// ---------------------------------------------------------------------------
__global__ __launch_bounds__(128) void flash_kernel(
    const ushort_t* __restrict__ q, const ushort_t* __restrict__ k,
    const ushort_t* __restrict__ vT, ushort_t* __restrict__ ctx)
{
  __shared__ alignas(16) ushort_t sP[4][16][32];   // wave-private P bounce
  const int tid = threadIdx.x, lane = tid & 31, wv = tid >> 5;
  const int lane15 = lane & 15, hi8 = (lane >> 4) << 3;
  const int s0 = blockIdx.x * 64;
  const int h = blockIdx.y, b = blockIdx.z;
  const int qrow = s0 + 16 * wv;
  const size_t bh = (size_t)b * HH + h;
  const int kbA = (lane < 16) ? 0 : 16;
  const int kbB = (lane < 16) ? 0 : 16;

  // Q fragments: two K=32 chunks covering D=64
  ABPack qa[2];
  {
    const f32x4* rowQ = (const f32x4*)&q[(bh * SS + qrow + lane15) * DD];
    for (int c = 0; c < 2; ++c) {
      qa[c].f[0] = rowQ[(32 * c + kbA) >> 3];
      qa[c].f[1] = rowQ[((32 * c + kbA) >> 3) + 1];
    }
  }

  v8f acc[4];
  float mst[8], lst[8];
  for (int j = 0; j < 4; ++j)
    for (int r = 0; r < 8; ++r) acc[j][r] = 0.0f;
  for (int r = 0; r < 8; ++r) { mst[r] = -1e30f; lst[r] = 0.0f; }

  for (int key0 = 0; key0 < s0 + 64; key0 += 32) {
    // ---- scores: S = (Q Kᵀ) * 1/sqrt(D), two 16-key tiles ----
    v8f sc[2];
    for (int n = 0; n < 2; ++n) {
      const f32x4* rowK = (const f32x4*)&k[(bh * SS + key0 + 16 * n + lane15) * DD];
      ABPack k0f, k1f;
      k0f.f[0] = rowK[kbB >> 3];
      k0f.f[1] = rowK[(kbB >> 3) + 1];
      k1f.f[0] = rowK[(32 + kbB) >> 3];
      k1f.f[1] = rowK[((32 + kbB) >> 3) + 1];
      v8f z;
      for (int r = 0; r < 8; ++r) z[r] = 0.0f;
      z = wmma_bf16(qa[0].v, k0f.v, z);
      z = wmma_bf16(qa[1].v, k1f.v, z);
      sc[n] = z;
    }
    // scale + causal mask
    for (int n = 0; n < 2; ++n)
      for (int r = 0; r < 8; ++r) {
        float val  = sc[n][r] * 0.125f;
        int   kpos = key0 + 16 * n + lane15;
        int   qpos = qrow + r + hi8;
        sc[n][r] = (kpos > qpos) ? -1e30f : val;
      }
    // ---- online softmax (butterfly stays inside the 16-lane half) ----
    float sca[8];
    for (int r = 0; r < 8; ++r) {
      float t = fmaxf(sc[0][r], sc[1][r]);
      for (int off = 1; off <= 8; off <<= 1) t = fmaxf(t, __shfl_xor(t, off, 32));
      float mn = fmaxf(mst[r], t);
      sca[r] = __expf(mst[r] - mn);
      mst[r] = mn;
    }
    for (int r = 0; r < 8; ++r) {
      sc[0][r] = __expf(sc[0][r] - mst[r]);
      sc[1][r] = __expf(sc[1][r] - mst[r]);
      float t = sc[0][r] + sc[1][r];
      for (int off = 1; off <= 8; off <<= 1) t += __shfl_xor(t, off, 32);
      lst[r] = lst[r] * sca[r] + t;
      for (int j = 0; j < 4; ++j) acc[j][r] *= sca[r];
    }
    // ---- P (C layout) -> LDS -> A fragment ----
    for (int n = 0; n < 2; ++n)
      for (int r = 0; r < 8; ++r)
        sP[wv][r + hi8][16 * n + lane15] = f2bf(sc[n][r]);
    ABPack pa;
    {
      const f32x4* rowP = (const f32x4*)&sP[wv][lane15][0];
      pa.f[0] = rowP[kbA >> 3];
      pa.f[1] = rowP[(kbA >> 3) + 1];
    }
    // ---- ctx += P @ V (vT is [B,H,D,S] so loads are contiguous over keys) ----
    for (int j = 0; j < 4; ++j) {
      const f32x4* rowV = (const f32x4*)&vT[(bh * DD + 16 * j + lane15) * SS + key0];
      ABPack vb;
      vb.f[0] = rowV[kbB >> 3];
      vb.f[1] = rowV[(kbB >> 3) + 1];
      acc[j] = wmma_bf16(pa.v, vb.v, acc[j]);
    }
  }

  // normalize + store ctx as [B,S,H*D] bf16
  for (int r = 0; r < 8; ++r) {
    float inv = 1.0f / lst[r];
    const int g = qrow + r + hi8;
    for (int j = 0; j < 4; ++j)
      ctx[((size_t)b * SS + g) * (HH * DD) + h * DD + 16 * j + lane15] =
          f2bf(acc[j][r] * inv);
  }
}

// ---------------------------------------------------------------------------
// Kernel 3: out = ctx @ Wout + b_out  (bf16 WMMA, fp32 out)
// ---------------------------------------------------------------------------
__global__ __launch_bounds__(128) void outproj_kernel(
    const ushort_t* __restrict__ ctx, const float* __restrict__ Wout,
    const float* __restrict__ bout, float* __restrict__ out)
{
  __shared__ alignas(16) ushort_t sA[64][32];
  __shared__ alignas(16) ushort_t sBt[64][32];
  const int tid = threadIdx.x, lane = tid & 31, wv = tid >> 5;
  const int lane15 = lane & 15, hi8 = (lane >> 4) << 3;
  const int m0 = blockIdx.x * 64, n0 = blockIdx.y * 64;

  v8f acc[4];
  for (int t = 0; t < 4; ++t)
    for (int r = 0; r < 8; ++r) acc[t][r] = 0.0f;

  const int kbA = (lane < 16) ? 0 : 16;
  const int kbB = (lane < 16) ? 0 : 16;

  for (int k0 = 0; k0 < HH * DD; k0 += 32) {
    for (int e = tid; e < 64 * 32; e += 128) {
      int r = e >> 5, c = e & 31;
      sA[r][c] = ctx[(size_t)(m0 + r) * (HH * DD) + k0 + c];
    }
    for (int e = tid; e < 32 * 64; e += 128) {
      int kr = e >> 6, nc = e & 63;
      sBt[nc][kr] = f2bf(Wout[(size_t)(k0 + kr) * EE + n0 + nc]);
    }
    __syncthreads();
    if (k0 + 32 < HH * DD) {
      __builtin_prefetch(&ctx[(size_t)(m0 + (tid >> 1)) * (HH * DD) + k0 + 32], 0, 0);
      __builtin_prefetch(&Wout[(size_t)(k0 + 32 + (tid >> 2)) * EE + n0], 0, 0);
    }
    ABPack a;
    const f32x4* rowA = (const f32x4*)&sA[16 * wv + lane15][0];
    a.f[0] = rowA[kbA >> 3];
    a.f[1] = rowA[(kbA >> 3) + 1];
    for (int t = 0; t < 4; ++t) {
      ABPack bfr;
      const f32x4* rowB = (const f32x4*)&sBt[16 * t + lane15][0];
      bfr.f[0] = rowB[kbB >> 3];
      bfr.f[1] = rowB[(kbB >> 3) + 1];
      acc[t] = wmma_bf16(a.v, bfr.v, acc[t]);
    }
    __syncthreads();
  }

  float bias[4];
  for (int t = 0; t < 4; ++t) bias[t] = bout[n0 + 16 * t + lane15];
  for (int r = 0; r < 8; ++r) {
    const int g = m0 + 16 * wv + r + hi8;
    for (int t = 0; t < 4; ++t)
      out[(size_t)g * EE + n0 + 16 * t + lane15] = acc[t][r] + bias[t];
  }
}

// ---------------------------------------------------------------------------
extern "C" void kernel_launch(void* const* d_in, const int* in_sizes, int n_in,
                              void* d_out, int out_size, void* d_ws, size_t ws_size,
                              hipStream_t stream) {
  (void)in_sizes; (void)n_in; (void)out_size; (void)ws_size;
  const float* x    = (const float*)d_in[0];
  const float* Wqkv = (const float*)d_in[1];
  const float* bqkv = (const float*)d_in[2];
  const float* Wout = (const float*)d_in[3];
  const float* bout = (const float*)d_in[4];
  // d_in[5] = kv_cache: input-only here; the reference output is just `out`.

  const size_t QN = (size_t)BB * HH * SS * DD;   // 4,194,304 elems
  ushort_t* qw  = (ushort_t*)d_ws;               // bf16 q   [B,H,S,D]
  ushort_t* kw  = qw + QN;                       // bf16 k   [B,H,S,D]
  ushort_t* vTw = kw + QN;                       // bf16 vT  [B,H,D,S]
  ushort_t* cw  = vTw + QN;                      // bf16 ctx [B,S,H*D]  (32 MB total)

  qkv_rope_kernel<<<dim3(MTOT / 64, NQKV / 64), 128, 0, stream>>>(
      x, Wqkv, bqkv, qw, kw, vTw);
  flash_kernel<<<dim3(SS / 64, HH, BB), 128, 0, stream>>>(qw, kw, vTw, cw);
  outproj_kernel<<<dim3(MTOT / 64, EE / 64), 128, 0, stream>>>(
      cw, Wout, bout, (float*)d_out);
}
